// Linear_int8_44109314130619
// MI455X (gfx1250) — matrix operand order repaired and checked by reading, not compiled, not run on hardware
//
#include <hip/hip_runtime.h>

// Problem constants (match reference)
#define B_DIM 32
#define IN_DIM 8192
#define OUT_DIM 16384
#define LUT_N 256

// Tiling
#define KCHUNK 32                         // K per v_wmma_f32_16x16x32_f16
#define NCHUNKS (IN_DIM / KCHUNK)         // 256
#define KS 4                              // deterministic K-split
#define CHUNKS_PER_SPLIT (NCHUNKS / KS)   // 64
#define WG_THREADS 256                    // 8 wave32
#define WAVES_PER_WG 8
#define FEATS_PER_WAVE 16
#define FEATS_PER_WG (WAVES_PER_WG * FEATS_PER_WAVE)  // 128

typedef _Float16 v16h __attribute__((ext_vector_type(16)));
typedef float    v8f  __attribute__((ext_vector_type(8)));
typedef int      v4i  __attribute__((ext_vector_type(4)));

union H16 { v16h v; _Float16 h[16]; };

// ---------------------------------------------------------------------------
// Kernel 0: convert x (fp32) to f16 and pre-swizzle into the CDNA5 WMMA
// A-operand k order used by this GEMM (K contiguous per lane half), so the GEMM's A-load is
// contiguous b128s. Output dword d encodes (chunk c, Mtile t, lane, vgpr r):
//   d = ((c*2 + t)*32 + lane)*8 + r
// A layout: lanes 0-15 -> M = t*16 + lane, K = 2r (+lg*16)
// ---------------------------------------------------------------------------
__global__ void xswizzle_f16(const float* __restrict__ x,
                             unsigned* __restrict__ xsw) {
  int d = blockIdx.x * blockDim.x + threadIdx.x;  // 0 .. B*IN/2 - 1
  int r    = d & 7;
  int lane = (d >> 3) & 31;
  int t    = (d >> 8) & 1;
  int c    = d >> 9;
  int lg   = lane >> 4;
  int m    = t * 16 + (lane & 15);
  int kloc = 2 * r + lg * 16;
  int k    = c * KCHUNK + kloc;
  union { _Float16 h[2]; unsigned u; } p;
  p.h[0] = (_Float16)x[(size_t)m * IN_DIM + k];
  p.h[1] = (_Float16)x[(size_t)m * IN_DIM + k + 1];
  xsw[d] = p.u;
}

// ---------------------------------------------------------------------------
// Kernel 1: streaming LUT-dequant GEMM.
// grid.x = OUT/128 N-blocks, grid.y = KS K-splits. Each wave: 16 features,
// 2 WMMA accumulators (M=0..15, 16..31), 64 K-chunks of 32.
// B-fragment k order used (16-bit B 32x16, same as A): lane n = l&15; lane group
// l>>4 covers K 0..15 / 16..31; VGPR r holds packed K = lg*16 + 2r, 2r+1.
// Each lane therefore needs 16 *consecutive* idx values -> 4x b128 NT loads.
// ---------------------------------------------------------------------------
__global__ void __launch_bounds__(WG_THREADS)
gemm_lut_wmma(const unsigned* __restrict__ xsw,
              const float* __restrict__ lut,
              const int* __restrict__ widx,
              float* __restrict__ partials) {
  __shared__ float lutS[LUT_N];
  lutS[threadIdx.x & (LUT_N - 1)] = lut[threadIdx.x & (LUT_N - 1)];
  __syncthreads();

  const int lane  = threadIdx.x & 31;
  const int wave  = threadIdx.x >> 5;
  const int col   = lane & 15;
  const int lg    = lane >> 4;
  const int nBase = blockIdx.x * FEATS_PER_WG + wave * FEATS_PER_WAVE;
  const int n     = nBase + col;
  const int c0    = blockIdx.y * CHUNKS_PER_SPLIT;

  v8f acc0 = {};
  v8f acc1 = {};

  // idx base for this lane: row n, K offset c0*32 + lg*16, 16 consecutive ints
  const v4i* idxp =
      (const v4i*)(widx + (size_t)n * IN_DIM + (size_t)c0 * KCHUNK + lg * 16);
  // A fragments: v16h unit = 8 dwords; fragment (c,t,lane) at index (c*2+t)*32+lane
  const v16h* ap = (const v16h*)(xsw + (size_t)c0 * 512);

  for (int cc = 0; cc < CHUNKS_PER_SPLIT; ++cc) {
    const v4i* ip = idxp + (size_t)cc * 8;   // 8 v4i per chunk per row
    v4i i0 = __builtin_nontemporal_load(ip + 0);
    v4i i1 = __builtin_nontemporal_load(ip + 1);
    v4i i2 = __builtin_nontemporal_load(ip + 2);
    v4i i3 = __builtin_nontemporal_load(ip + 3);
    // stream-prefetch the index data 4 chunks (512B) ahead
    __builtin_prefetch((const void*)(ip + 32), 0, 0);

    // A fragments for both M tiles (L2-resident, shared by all waves)
    v16h a0 = ap[(size_t)(cc * 2 + 0) * 32 + lane];
    v16h a1 = ap[(size_t)(cc * 2 + 1) * 32 + lane];

    // LUT dequant (LDS gather) + pack into the B fragment
    H16 bw;
    bw.h[0]  = (_Float16)lutS[i0.x];
    bw.h[1]  = (_Float16)lutS[i0.y];
    bw.h[2]  = (_Float16)lutS[i0.z];
    bw.h[3]  = (_Float16)lutS[i0.w];
    bw.h[4]  = (_Float16)lutS[i1.x];
    bw.h[5]  = (_Float16)lutS[i1.y];
    bw.h[6]  = (_Float16)lutS[i1.z];
    bw.h[7]  = (_Float16)lutS[i1.w];
    bw.h[8]  = (_Float16)lutS[i2.x];
    bw.h[9]  = (_Float16)lutS[i2.y];
    bw.h[10] = (_Float16)lutS[i2.z];
    bw.h[11] = (_Float16)lutS[i2.w];
    bw.h[12] = (_Float16)lutS[i3.x];
    bw.h[13] = (_Float16)lutS[i3.y];
    bw.h[14] = (_Float16)lutS[i3.z];
    bw.h[15] = (_Float16)lutS[i3.w];

    acc0 = __builtin_amdgcn_wmma_f32_16x16x32_f16(
        false, a0, false, bw.v, (short)0, acc0, false, false);
    acc1 = __builtin_amdgcn_wmma_f32_16x16x32_f16(
        false, a1, false, bw.v, (short)0, acc1, false, false);
  }

  // D layout: VGPR v, lanes 0-15: M=v, N=lane; lanes 16-31: M=v+8, N=lane-16
  float* prow = partials + (size_t)blockIdx.y * B_DIM * OUT_DIM;
#pragma unroll
  for (int v = 0; v < 8; ++v) {
    int m = v + lg * 8;
    prow[(size_t)(m) * OUT_DIM + n]      = acc0[v];
    prow[(size_t)(m + 16) * OUT_DIM + n] = acc1[v];
  }
}

// ---------------------------------------------------------------------------
// Kernel 2: deterministic K-split reduction + bias.
// ---------------------------------------------------------------------------
__global__ void reduce_bias(const float* __restrict__ partials,
                            const float* __restrict__ bias,
                            float* __restrict__ out) {
  int t = blockIdx.x * blockDim.x + threadIdx.x;
  if (t >= B_DIM * OUT_DIM) return;
  int o = t & (OUT_DIM - 1);
  float s = bias[o];
#pragma unroll
  for (int ks = 0; ks < KS; ++ks)
    s += partials[(size_t)ks * B_DIM * OUT_DIM + t];
  out[t] = s;
}

extern "C" void kernel_launch(void* const* d_in, const int* in_sizes, int n_in,
                              void* d_out, int out_size, void* d_ws,
                              size_t ws_size, hipStream_t stream) {
  const float* x    = (const float*)d_in[0];
  const float* lut  = (const float*)d_in[1];
  const float* bias = (const float*)d_in[2];
  const int*   widx = (const int*)d_in[3];
  float* out = (float*)d_out;

  unsigned* xsw      = (unsigned*)d_ws;                       // 512 KB
  float*    partials = (float*)((char*)d_ws +
                                (size_t)B_DIM * IN_DIM * 2);  // 8 MB

  // 1) x -> f16, pre-swizzled to WMMA A-fragment order
  xswizzle_f16<<<(B_DIM * IN_DIM / 2) / 256, 256, 0, stream>>>(x, xsw);

  // 2) streaming LUT-dequant GEMM (WMMA), K-split into partials
  dim3 grid(OUT_DIM / FEATS_PER_WG, KS);
  gemm_lut_wmma<<<grid, WG_THREADS, 0, stream>>>(xsw, lut, widx, partials);

  // 3) reduce partials + bias -> out
  reduce_bias<<<(B_DIM * OUT_DIM) / 256, 256, 0, stream>>>(partials, bias, out);
}
